// GNNEncoder_25331717112514
// MI455X (gfx1250) — matrix operand order repaired and checked by reading, not compiled, not run on hardware
//
#include <hip/hip_runtime.h>
#include <hip/hip_bf16.h>

#define NUM_GRAPHS 128
#define NPG 512
#define NN (NUM_GRAPHS * NPG)   // 65536 nodes
#define NE (NN * 16)            // 1048576 edges
#define IN_DIM 64
#define HID 128
#define NHEADS 4
#define DH 32
#define QKVW 384

typedef __attribute__((ext_vector_type(16))) _Float16 v16h;
typedef __attribute__((ext_vector_type(8)))  _Float16 v8h;
typedef __attribute__((ext_vector_type(8)))  float    v8f;

__device__ __forceinline__ v8f wmma16(v16h a, v16h b, v8f c) {
  // D = A(16x32 f16) * B(32x16 f16) + C(16x16 f32)
  return __builtin_amdgcn_wmma_f32_16x16x32_f16(false, a, false, b, (short)0, c,
                                                false, false);
}

// A fragment from f16 source: lane L holds row m=L%16; halves cover
// K = g*8+{0..7} and 16+g*8+{0..7}, g = L/16 (ISA 7.12.2).
__device__ __forceinline__ v16h load_a_frag(const _Float16* base, int ld, int lane) {
  int m = lane & 15;
  int g = lane >> 4;
  const _Float16* p = base + m * ld + g * 8;
  v8h lo = *(const v8h*)p;          // K = g*8 .. g*8+7
  v8h hi = *(const v8h*)(p + 16);   // K = 16+g*8 .. 16+g*8+7
  return __builtin_shufflevector(lo, hi, 0,1,2,3,4,5,6,7,8,9,10,11,12,13,14,15);
}

// Same fragment but from an f32 source (LDS tile staged by async DMA),
// converting to f16 while building the fragment.
__device__ __forceinline__ v16h load_a_frag_f32(const float* base, int ld, int lane) {
  int m = lane & 15;
  int g = lane >> 4;
  const float* p = base + m * ld + g * 8;
  v16h a;
#pragma unroll
  for (int i = 0; i < 8; i++) a[i] = (_Float16)p[i];
#pragma unroll
  for (int i = 0; i < 8; i++) a[i + 8] = (_Float16)p[16 + i];
  return a;
}

// B fragment: lane L holds column n=L%16; halves cover K = g*8+{0..7} and 16+g*8+{0..7}, g = L/16. base at [k=0,n=0], stride ld.
__device__ __forceinline__ v16h load_b_frag(const _Float16* base, int ld, int lane) {
  int n = lane & 15;
  int g = lane >> 4;
  v16h b;
#pragma unroll
  for (int i = 0; i < 8; i++) b[i] = base[(long)(g * 8 + i) * ld + n];
#pragma unroll
  for (int i = 0; i < 8; i++) b[i + 8] = base[(long)(16 + g * 8 + i) * ld + n];
  return b;
}

// ---------------------------------------------------------------- degree / norm
__global__ void init_deg_kernel(float* deg) {
  int i = blockIdx.x * blockDim.x + threadIdx.x;
  if (i < NN) deg[i] = 1.0f;   // self-loop
}
__global__ void accum_deg_kernel(const int* __restrict__ dstE, float* deg) {
  int e = blockIdx.x * blockDim.x + threadIdx.x;
  if (e < NE) unsafeAtomicAdd(&deg[dstE[e]], 1.0f);
}
__global__ void make_dis_kernel(const float* __restrict__ deg, float* dis) {
  int i = blockIdx.x * blockDim.x + threadIdx.x;
  if (i < NN) dis[i] = rsqrtf(deg[i]);
}

// ------------------------------------------------- aggregation: z = selfnorm*h + scatter
__global__ void self_term_kernel(const float* __restrict__ h, const float* __restrict__ dis,
                                 float* __restrict__ z) {
  int node = blockIdx.x;
  int f = threadIdx.x;
  int F = blockDim.x;
  float s = dis[node];
  z[(long)node * F + f] = h[(long)node * F + f] * s * s;
}
__global__ void edge_scatter_kernel(const float* __restrict__ h, const int* __restrict__ srcE,
                                    const int* __restrict__ dstE, const float* __restrict__ dis,
                                    float* __restrict__ z) {
  int e = blockIdx.x;
  int f = threadIdx.x;
  int F = blockDim.x;
  int s = srcE[e];
  int d = dstE[e];
  float w = dis[s] * dis[d];
  unsafeAtomicAdd(&z[(long)d * F + f], h[(long)s * F + f] * w);
}

// ------------------------------------------------- generic WMMA GEMM + alpha/beta + relu
// C[m,n] = relu?( (sum_k A[m,k]*B[k,n]) * alpha[n] + beta[n] )
// A tile (contiguous 16*K f32 block, lda==K) is staged into LDS with
// GLOBAL_LOAD_ASYNC_TO_LDS_B128 (ASYNCcnt path), then converted to f16 per-fragment.
__global__ void gemm_bn_relu_kernel(const float* __restrict__ A, int kbits,
                                    const _Float16* __restrict__ B, int Nn,
                                    const float* __restrict__ alpha,
                                    const float* __restrict__ beta,
                                    int relu, float* __restrict__ C,
                                    _Float16* __restrict__ C16) {
  __shared__ __align__(16) float sAraw[16 * 128];   // 8 KB
  const int K = 1 << kbits;
  const int m0 = blockIdx.x << 4;
  const int tid = threadIdx.y * 32 + threadIdx.x;
  const int nthr = blockDim.y * 32;

  // async DMA the contiguous 16xK f32 tile into LDS, 16 B per lane per iteration
  {
    unsigned ldsBase = (unsigned)(size_t)(void*)sAraw;   // LDS offset = low 32 bits
    const float* gbase = A + ((long)m0 << kbits);
    const int nchunks = (16 << kbits) >> 2;              // # of 16-byte chunks
    for (int i = tid; i < nchunks; i += nthr) {
      unsigned loff = ldsBase + i * 16;
      const float* gp = gbase + i * 4;
      asm volatile("global_load_async_to_lds_b128 %0, %1, off"
                   :: "v"(loff), "v"(gp)
                   : "memory");
    }
    asm volatile("s_wait_asynccnt 0x0" ::: "memory");
  }
  __syncthreads();

  const int lane = threadIdx.x;
  const int n0 = threadIdx.y << 4;
  v8f acc = {};
  for (int k0 = 0; k0 < K; k0 += 32) {
    v16h a = load_a_frag_f32(sAraw + k0, K, lane);
    v16h b = load_b_frag(B + (long)k0 * Nn + n0, Nn, lane);
    acc = wmma16(a, b, acc);
  }
  const int nloc = lane & 15;
  const int mh = (lane >> 4) * 8;
  const int n = n0 + nloc;
  const float al = alpha[n];
  const float be = beta[n];
#pragma unroll
  for (int r = 0; r < 8; r++) {
    float y = acc[r] * al + be;
    if (relu) y = fmaxf(y, 0.0f);
    long idx = (long)(m0 + mh + r) * Nn + n;
    if (C)   C[idx] = y;
    if (C16) C16[idx] = (_Float16)y;
  }
}

// ------------------------------------------------- weight prep (f16 convert / transpose / BN fold)
__global__ void prep_weights_kernel(const float* W0, const float* b0, const float* Wh,
                                    const float* bh, const float* gam, const float* bet,
                                    const float* mean, const float* var, const float* Win,
                                    const float* binq, const float* Wout, const float* bout,
                                    _Float16* W0h, _Float16* Whh, _Float16* WinT,
                                    _Float16* WoutT, float* alphaG, float* betaG,
                                    float* alphaQ, float* betaQ, float* alphaO, float* betaO) {
  int tid = blockIdx.x * blockDim.x + threadIdx.x;
  int stride = gridDim.x * blockDim.x;
  for (int i = tid; i < IN_DIM * HID; i += stride) W0h[i] = (_Float16)W0[i];
  for (int i = tid; i < 3 * HID * HID; i += stride) Whh[i] = (_Float16)Wh[i];
  for (int i = tid; i < HID * QKVW; i += stride) {
    int k = i / QKVW, j = i - k * QKVW;
    WinT[i] = (_Float16)Win[j * HID + k];   // qkv = f @ Win^T
  }
  for (int i = tid; i < HID * HID; i += stride) {
    int k = i >> 7, j = i & 127;
    WoutT[i] = (_Float16)Wout[(j << 7) + k];
  }
  for (int i = tid; i < 4 * HID; i += stride) {
    int l = i >> 7, c = i & 127;
    float sc = gam[i] * rsqrtf(var[i] + 1e-5f);
    float bb = (l == 0) ? b0[c] : bh[(l - 1) * HID + c];
    alphaG[i] = sc;
    betaG[i] = (bb - mean[i]) * sc + bet[i];
  }
  for (int i = tid; i < QKVW; i += stride) { alphaQ[i] = 1.0f; betaQ[i] = binq[i]; }
  for (int i = tid; i < HID; i += stride)  { alphaO[i] = 1.0f; betaO[i] = bout[i]; }
}

// ------------------------------------------------- repack K into [g][h][d][key] (f16)
__global__ void repack_kT_kernel(const _Float16* __restrict__ qkv, _Float16* __restrict__ kT) {
  int i = blockIdx.x * blockDim.x + threadIdx.x;  // < NN*HID
  int node = i >> 7;
  int c = i & 127;                  // c = h*32 + d
  int g = node >> 9;
  int n = node & 511;
  kT[(((long)g * HID + c) << 9) + n] = qkv[(long)node * QKVW + HID + c];
}

// ------------------------------------------------- fused attention per (graph, head, row-tile)
__global__ void attn_kernel(const _Float16* __restrict__ qkv,
                            const _Float16* __restrict__ kT, float* __restrict__ o) {
  __shared__ float    sS[16 * NPG];   // scores f32 (32 KB)
  __shared__ _Float16 sP[16 * NPG];   // probs  f16 (16 KB)
  const int t = blockIdx.x;   // row tile (0..31)
  const int h = blockIdx.y;   // head
  const int g = blockIdx.z;   // graph
  const int lane = threadIdx.x;
  const float scale = 0.1767766952966369f;  // 1/sqrt(32)
  const int nloc = lane & 15;
  const int mh = (lane >> 4) * 8;

  // Q fragment (16 rows x 32 d), reused across all 32 score WMMAs
  const _Float16* qbase = qkv + ((long)(g * NPG + t * 16)) * QKVW + h * DH;
  v16h aq = load_a_frag(qbase, QKVW, lane);

  const _Float16* kTb = kT + ((long)((g * NHEADS + h) * DH)) * NPG;  // [d][key]
  for (int j0 = 0; j0 < NPG; j0 += 16) {
    v16h bk = load_b_frag(kTb + j0, NPG, lane);  // lane=key, halves=d
    v8f c = {};
    c = wmma16(aq, bk, c);
#pragma unroll
    for (int r = 0; r < 8; r++)
      sS[(mh + r) * NPG + j0 + nloc] = c[r] * scale;
  }
  __syncthreads();

  // softmax: two lanes per row, 256 columns each
  {
    const int row = lane >> 1;
    const int c0 = (lane & 1) * (NPG / 2);
    float mx = -3.0e38f;
    for (int c = 0; c < NPG / 2; c++) mx = fmaxf(mx, sS[row * NPG + c0 + c]);
    mx = fmaxf(mx, __shfl_xor(mx, 1, 32));
    float sum = 0.0f;
    for (int c = 0; c < NPG / 2; c++) {
      float e = __expf(sS[row * NPG + c0 + c] - mx);
      sS[row * NPG + c0 + c] = e;
      sum += e;
    }
    sum += __shfl_xor(sum, 1, 32);
    float inv = 1.0f / sum;
    for (int c = 0; c < NPG / 2; c++)
      sP[row * NPG + c0 + c] = (_Float16)(sS[row * NPG + c0 + c] * inv);
  }
  __syncthreads();

  // O = P(16x512) @ V(512x32), V read from qkv (contiguous in d)
  const _Float16* vbase = qkv + ((long)(g * NPG)) * QKVW + 2 * HID + h * DH;
  for (int n0 = 0; n0 < DH; n0 += 16) {
    v8f c = {};
    for (int k0 = 0; k0 < NPG; k0 += 32) {
      v16h ap = load_a_frag(sP + k0, NPG, lane);
      v16h bv = load_b_frag(vbase + (long)k0 * QKVW + n0, QKVW, lane);
      c = wmma16(ap, bv, c);
    }
#pragma unroll
    for (int r = 0; r < 8; r++) {
      int mrow = t * 16 + mh + r;
      o[((long)(g * NPG + mrow)) * HID + h * DH + n0 + nloc] = c[r];
    }
  }
}

// ------------------------------------------------- mean pool per graph
__global__ void pool_kernel(const float* __restrict__ fin, float* __restrict__ emb) {
  int g = blockIdx.x;
  int c = threadIdx.x;
  float s = 0.0f;
  for (int n = 0; n < NPG; n++) s += fin[((long)(g * NPG + n)) * HID + c];
  emb[g * HID + c] = s * (1.0f / NPG);
}

extern "C" void kernel_launch(void* const* d_in, const int* in_sizes, int n_in,
                              void* d_out, int out_size, void* d_ws, size_t ws_size,
                              hipStream_t stream) {
  const float* x    = (const float*)d_in[0];
  const float* W0   = (const float*)d_in[1];
  const float* b0   = (const float*)d_in[2];
  const float* Wh   = (const float*)d_in[3];
  const float* bh   = (const float*)d_in[4];
  const float* gam  = (const float*)d_in[5];
  const float* bet  = (const float*)d_in[6];
  const float* mean = (const float*)d_in[7];
  const float* var  = (const float*)d_in[8];
  const float* Win  = (const float*)d_in[9];
  const float* binq = (const float*)d_in[10];
  const float* Wout = (const float*)d_in[11];
  const float* bout = (const float*)d_in[12];
  const int* eidx   = (const int*)d_in[13];
  const int* srcE = eidx;
  const int* dstE = eidx + NE;

  char* p = (char*)d_ws;
  auto carve = [&](size_t bytes) {
    char* r = p;
    p += (bytes + 255) & ~size_t(255);
    return r;
  };
  float* deg        = (float*)carve((size_t)NN * 4);
  float* dis        = (float*)carve((size_t)NN * 4);
  float* zbuf       = (float*)carve((size_t)NN * HID * 4);
  float* hbuf       = (float*)carve((size_t)NN * HID * 4);
  _Float16* qkv16   = (_Float16*)carve((size_t)NN * QKVW * 2);
  _Float16* kT      = (_Float16*)carve((size_t)NN * HID * 2);
  _Float16* W0h     = (_Float16*)carve((size_t)IN_DIM * HID * 2);
  _Float16* Whh     = (_Float16*)carve((size_t)3 * HID * HID * 2);
  _Float16* WinT    = (_Float16*)carve((size_t)HID * QKVW * 2);
  _Float16* WoutT   = (_Float16*)carve((size_t)HID * HID * 2);
  float* alphaG     = (float*)carve(4 * HID * 4);
  float* betaG      = (float*)carve(4 * HID * 4);
  float* alphaQ     = (float*)carve(QKVW * 4);
  float* betaQ      = (float*)carve(QKVW * 4);
  float* alphaO     = (float*)carve(HID * 4);
  float* betaO      = (float*)carve(HID * 4);

  float* finalOut = (float*)d_out;
  float* embOut = finalOut + (size_t)NN * HID;

  // degree / normalization
  init_deg_kernel<<<NN / 256, 256, 0, stream>>>(deg);
  accum_deg_kernel<<<NE / 256, 256, 0, stream>>>(dstE, deg);
  make_dis_kernel<<<NN / 256, 256, 0, stream>>>(deg, dis);

  // weight prep
  prep_weights_kernel<<<96, 256, 0, stream>>>(W0, b0, Wh, bh, gam, bet, mean, var, Win, binq,
                                              Wout, bout, W0h, Whh, WinT, WoutT, alphaG, betaG,
                                              alphaQ, betaQ, alphaO, betaO);

  // layer 0: aggregate in IN_DIM=64 space (GCN is linear -> aggregate before matmul)
  self_term_kernel<<<NN, IN_DIM, 0, stream>>>(x, dis, zbuf);
  edge_scatter_kernel<<<NE, IN_DIM, 0, stream>>>(x, srcE, dstE, dis, zbuf);
  gemm_bn_relu_kernel<<<NN / 16, dim3(32, 8), 0, stream>>>(zbuf, 6, W0h, HID, alphaG, betaG, 1,
                                                           hbuf, nullptr);
  // layers 1..3
  for (int l = 1; l <= 3; ++l) {
    self_term_kernel<<<NN, HID, 0, stream>>>(hbuf, dis, zbuf);
    edge_scatter_kernel<<<NE, HID, 0, stream>>>(hbuf, srcE, dstE, dis, zbuf);
    gemm_bn_relu_kernel<<<NN / 16, dim3(32, 8), 0, stream>>>(
        zbuf, 7, Whh + (size_t)(l - 1) * HID * HID, HID, alphaG + l * HID, betaG + l * HID, 1,
        hbuf, nullptr);
  }

  // QKV projection -> f16 buffer
  gemm_bn_relu_kernel<<<NN / 16, dim3(32, 24), 0, stream>>>(hbuf, 7, WinT, QKVW, alphaQ, betaQ,
                                                            0, nullptr, qkv16);
  // repack K transposed per (graph, head)
  repack_kT_kernel<<<(NN * HID) / 256, 256, 0, stream>>>(qkv16, kT);

  // attention -> o into zbuf
  attn_kernel<<<dim3(NPG / 16, NHEADS, NUM_GRAPHS), 32, 0, stream>>>(qkv16, kT, zbuf);

  // output projection -> final, then pool
  gemm_bn_relu_kernel<<<NN / 16, dim3(32, 8), 0, stream>>>(zbuf, 7, WoutT, HID, alphaO, betaO, 0,
                                                           finalOut, nullptr);
  pool_kernel<<<NUM_GRAPHS, HID, 0, stream>>>(finalOut, embOut);
}
